// Mamba2_24584392802946
// MI455X (gfx1250) — compile-verified, matrix operand order checked by reading, not run on hardware
//
#include <hip/hip_runtime.h>

// ---------------- problem constants (match reference) ----------------
#define D_MODEL   1024
#define D_STATE   128
#define D_CONV    4
#define HEADDIM   64
#define CHUNK     256
#define D_INNER   2048
#define D_SSM     2048
#define NHEADS    32          // D_SSM / HEADDIM
#define CONV_DIM  (D_SSM + 2*D_STATE)          // 2304
#define D_IN_PROJ (2*D_INNER + 2*D_STATE + NHEADS) // 4384
#define D_OUT     1024
#define B_SZ      2
#define SEQ       4096
#define NC        (SEQ/CHUNK) // 16
#define EPS_RMS   1e-5f

// ---------------- WMMA fragment types / helpers ----------------
typedef __attribute__((ext_vector_type(16))) __bf16 v16bf;
typedef __attribute__((ext_vector_type(8)))  float  v8f;

// CDNA5 16-bit A/B fragment K-index for element i (0..15) given lane half:
// lanes 0-15 hold K {0..7, 16..23}; lanes 16-31 hold K {8..15, 24..31}.
__device__ __forceinline__ int kmap(int i, int hf) {
  return ((i >> 3) << 4) + (hf << 3) + (i & 7);
}

// Load a 16-element bf16 fragment from a row with contiguous K.
// Works for global or LDS (generic pointer) sources.
__device__ __forceinline__ v16bf frag_contig(const float* rowp, int k0, int hf) {
  v16bf r;
#pragma unroll
  for (int g = 0; g < 2; ++g) {
    const float* p = rowp + k0 + g * 16 + hf * 8;
#pragma unroll
    for (int j = 0; j < 8; ++j) r[g * 8 + j] = (__bf16)p[j];
  }
  return r;
}

__device__ __forceinline__ v8f wmma_bf16(v16bf a, v16bf b, v8f c) {
  return __builtin_amdgcn_wmma_f32_16x16x32_bf16(false, a, false, b, (short)0, c, false, false);
}

// ---------------- 1/4/9: LDS-staged batched GEMM  C = A * W^T ----------------
// A: [M,K] row-major (lda), W: [N,K] row-major (ldw) -> C: [M,N] (ldc)
// Block = 128 threads (4 waves). Block tile: 256(M) x 16(N).
// Per wave: 64(M) x 16(N) = 4 accumulators; B fragment loaded once per K-step
// from an LDS panel staged with global_load_async_to_lds_b128 (ASYNCcnt).
// grid: (N/16, M/256, batch)
#define KSLAB 512   // 16 x 512 fp32 = 32 KB LDS slab

__global__ void gemm_bf16_lds(const float* __restrict__ A, int lda, long long strideA,
                              const float* __restrict__ W, int ldw, long long strideW,
                              float* __restrict__ C, int ldc, long long strideC,
                              int K) {
  __shared__ __align__(16) float smemB[16 * KSLAB];
  const int tid = threadIdx.x;                 // 0..127
  const int wave = tid >> 5, lane = tid & 31;
  const int hf = lane >> 4, lane16 = lane & 15;
  const int n0 = blockIdx.x * 16;
  const int m0 = blockIdx.y * 256 + wave * 64;
  const float* Ab = A + (size_t)blockIdx.z * strideA;
  const float* Wb = W + (size_t)blockIdx.z * strideW + (size_t)n0 * ldw;
  const float* arow[4];
#pragma unroll
  for (int t = 0; t < 4; ++t) arow[t] = Ab + (size_t)(m0 + t * 16 + lane16) * lda;
  v8f acc[4] = {};

  for (int ks = 0; ks < K; ks += KSLAB) {
    const int klen = (K - ks) < KSLAB ? (K - ks) : KSLAB;
    // ---- stage B[16][klen] into LDS: async copy, drained by s_wait_asynccnt ----
    const int nvec = (16 * klen) >> 2;         // float4 transfers
    for (int idx = tid; idx < nvec; idx += 128) {
      const int row = idx / (klen >> 2);
      const int c4  = idx % (klen >> 2);
      const float* gsrc = Wb + (size_t)row * ldw + ks + c4 * 4;
      float* ldst = &smemB[row * KSLAB + c4 * 4];
#if defined(__gfx1250__)
      unsigned int loff = (unsigned int)(unsigned long long)ldst; // flat->LDS: addr[31:0]
      asm volatile("global_load_async_to_lds_b128 %0, %1, off"
                   :: "v"(loff), "v"(gsrc) : "memory");
#else
      ldst[0] = gsrc[0]; ldst[1] = gsrc[1]; ldst[2] = gsrc[2]; ldst[3] = gsrc[3];
#endif
    }
#if defined(__gfx1250__)
    asm volatile("s_wait_asynccnt 0x0" ::: "memory");
#endif
    __syncthreads();

    const float* brow = &smemB[lane16 * KSLAB];
    for (int k0 = 0; k0 < klen; k0 += 32) {
      __builtin_prefetch(arow[0] + ks + k0 + 256, 0, 1);  // global_prefetch_b8
      v16bf b = frag_contig(brow, k0, hf);                // ds_load path
#pragma unroll
      for (int t = 0; t < 4; ++t) {
        v16bf a = frag_contig(arow[t], ks + k0, hf);
        acc[t] = wmma_bf16(a, b, acc[t]);
      }
    }
    __syncthreads();
  }

  float* Cb = C + (size_t)blockIdx.z * strideC;
#pragma unroll
  for (int t = 0; t < 4; ++t)
#pragma unroll
    for (int r = 0; r < 8; ++r)
      Cb[(size_t)(m0 + t * 16 + hf * 8 + r) * ldc + n0 + lane16] = acc[t][r];
}

// ---------------- 2: depthwise causal conv (width 4) + SiLU + softplus(dt) ----------------
__global__ void conv_silu_dt(const float* __restrict__ zx,
                             const float* __restrict__ cw, const float* __restrict__ cb,
                             const float* __restrict__ dt_bias,
                             float* __restrict__ xBC, float* __restrict__ dtp) {
  const int bl = blockIdx.x;
  const int l = bl % SEQ, b = bl / SEQ;
  for (int c = threadIdx.x; c < CONV_DIM; c += blockDim.x) {
    float s = cb[c];
#pragma unroll
    for (int kk = 0; kk < D_CONV; ++kk) {
      int ls = l + kk - (D_CONV - 1);
      if (ls >= 0)
        s += zx[((size_t)b * SEQ + ls) * D_IN_PROJ + D_SSM + c] * cw[c * D_CONV + kk];
    }
    float sig = 1.f / (1.f + __expf(-s));
    xBC[(size_t)bl * CONV_DIM + c] = s * sig;
  }
  if (threadIdx.x < NHEADS) {
    int h = threadIdx.x;
    float x = zx[(size_t)bl * D_IN_PROJ + D_SSM + CONV_DIM + h] + dt_bias[h];
    dtp[(size_t)bl * NHEADS + h] = (x > 20.f) ? x : log1pf(__expf(x));
  }
}

// ---------------- 3: per-(b,h,chunk) inclusive cumsum of dA ----------------
__global__ void cumsum_dA(const float* __restrict__ dtp, const float* __restrict__ A_log,
                          float* __restrict__ cs) {
  __shared__ float sm[CHUNK];
  const int idx = blockIdx.x;
  const int z = idx % NC, h = (idx / NC) % NHEADS, b = idx / (NC * NHEADS);
  const int l = threadIdx.x;
  const float Ah = -__expf(A_log[h]);
  sm[l] = dtp[((size_t)b * SEQ + (size_t)z * CHUNK + l) * NHEADS + h] * Ah;
  __syncthreads();
  for (int off = 1; off < CHUNK; off <<= 1) {
    float t = (l >= off) ? sm[l - off] : 0.f;
    __syncthreads();
    sm[l] += t;
    __syncthreads();
  }
  cs[((size_t)(b * NHEADS + h) * SEQ) + (size_t)z * CHUNK + l] = sm[l];
}

// ---------------- 5: chunk states[p,n] = sum_l (x*dtp*decay)[l,p] * B[l,n] ----------------
// grid: (D_STATE/16, HEADDIM/16, B*NC*NHEADS); block 32
__global__ void states_kernel(const float* __restrict__ xBC, const float* __restrict__ dtp,
                              const float* __restrict__ cs, float* __restrict__ states) {
  __shared__ float w[CHUNK];
  const int batch = blockIdx.z;
  const int h = batch % NHEADS, z = (batch / NHEADS) % NC, b = batch / (NHEADS * NC);
  const size_t blbase = (size_t)b * SEQ + (size_t)z * CHUNK;
  const float* csrow = cs + ((size_t)(b * NHEADS + h) * SEQ) + (size_t)z * CHUNK;
  const float cslast = csrow[CHUNK - 1];
  for (int l = threadIdx.x; l < CHUNK; l += 32)
    w[l] = dtp[(blbase + l) * NHEADS + h] * __expf(cslast - csrow[l]);
  __syncthreads();
  const int lane = threadIdx.x & 31, hf = lane >> 4, lane16 = lane & 15;
  const int p0 = blockIdx.y * 16, n0 = blockIdx.x * 16;
  const float* xcol = xBC + blbase * CONV_DIM + h * HEADDIM + p0 + lane16;  // + l*CONV_DIM
  const float* bcol = xBC + blbase * CONV_DIM + D_SSM + n0 + lane16;        // + l*CONV_DIM
  v8f acc = {};
  for (int k0 = 0; k0 < CHUNK; k0 += 32) {
    v16bf a, bb;
#pragma unroll
    for (int i = 0; i < 16; ++i) {
      int l = k0 + kmap(i, hf);
      a[i]  = (__bf16)(xcol[(size_t)l * CONV_DIM] * w[l]);
      bb[i] = (__bf16)(bcol[(size_t)l * CONV_DIM]);
    }
    acc = wmma_bf16(a, bb, acc);
  }
  float* out = states + (size_t)batch * HEADDIM * D_STATE;
#pragma unroll
  for (int r = 0; r < 8; ++r)
    out[(size_t)(p0 + hf * 8 + r) * D_STATE + n0 + lane16] = acc[r];
}

// ---------------- 6: inter-chunk recurrence  prev[z] = carry before chunk z ----------------
__global__ void scan_prev(const float* __restrict__ states, const float* __restrict__ cs,
                          float* __restrict__ prev) {
  const size_t idx = (size_t)blockIdx.x * blockDim.x + threadIdx.x;
  const int n = idx % D_STATE, p = (idx / D_STATE) % HEADDIM;
  const int h = (idx / ((size_t)D_STATE * HEADDIM)) % NHEADS;
  const int b = idx / ((size_t)D_STATE * HEADDIM * NHEADS);
  float carry = 0.f;
  for (int z = 0; z < NC; ++z) {
    const size_t off = (((((size_t)b * NC + z) * NHEADS + h) * HEADDIM + p) * D_STATE) + n;
    prev[off] = carry;
    float cd = __expf(cs[((size_t)(b * NHEADS + h) * SEQ) + (size_t)z * CHUNK + CHUNK - 1]);
    carry = carry * cd + states[off];
  }
}

// ---------------- 7: Y = (L∘G) @ xd + (C·e^cs) @ prev^T + D*x ; t = Y * silu(z) ------------
// grid: (HEADDIM/16, CHUNK/16, B*NC*NHEADS); block 32
__global__ void ydiag_kernel(const float* __restrict__ xBC, const float* __restrict__ zx,
                             const float* __restrict__ dtp, const float* __restrict__ cs,
                             const float* __restrict__ G, const float* __restrict__ prev,
                             const float* __restrict__ Dp, float* __restrict__ t) {
  __shared__ float scs[CHUNK];
  __shared__ float sdt[CHUNK];
  const int batch = blockIdx.z;
  const int h = batch % NHEADS, z = (batch / NHEADS) % NC, b = batch / (NHEADS * NC);
  const size_t blbase = (size_t)b * SEQ + (size_t)z * CHUNK;
  const float* csrow = cs + ((size_t)(b * NHEADS + h) * SEQ) + (size_t)z * CHUNK;
  for (int l = threadIdx.x; l < CHUNK; l += 32) {
    scs[l] = csrow[l];
    sdt[l] = dtp[(blbase + l) * NHEADS + h];
  }
  __syncthreads();
  const int lane = threadIdx.x & 31, hf = lane >> 4, lane16 = lane & 15;
  const int l0 = blockIdx.y * 16, p0 = blockIdx.x * 16;
  const int lrow = l0 + lane16;
  const float cl = scs[lrow];
  const float* grow = G + ((size_t)(b * NC + z)) * CHUNK * CHUNK + (size_t)lrow * CHUNK;
  const float* xcol = xBC + blbase * CONV_DIM + h * HEADDIM + p0 + lane16;  // + s*CONV_DIM
  v8f acc = {};
  // --- diagonal block: K over s (within-chunk, causal masked, decay-weighted) ---
  for (int k0 = 0; k0 < CHUNK; k0 += 32) {
    v16bf a, bb;
#pragma unroll
    for (int i = 0; i < 16; ++i) {
      int s = k0 + kmap(i, hf);
      float av = (s <= lrow) ? grow[s] * __expf(cl - scs[s]) : 0.f;
      a[i]  = (__bf16)av;
      bb[i] = (__bf16)(xcol[(size_t)s * CONV_DIM] * sdt[s]);
    }
    acc = wmma_bf16(a, bb, acc);
  }
  // --- off-diagonal: K over n (state dim) with carried states ---
  const float el = __expf(cl);
  const float* crow = xBC + (blbase + lrow) * CONV_DIM + D_SSM + D_STATE;       // C[lrow, n]
  const float* prow = prev + ((size_t)batch * HEADDIM + p0 + lane16) * D_STATE; // prev[p, n]
  for (int k0 = 0; k0 < D_STATE; k0 += 32) {
    v16bf a, bb;
#pragma unroll
    for (int i = 0; i < 16; ++i) {
      int n = k0 + kmap(i, hf);
      a[i]  = (__bf16)(crow[n] * el);
      bb[i] = (__bf16)prow[n];
    }
    acc = wmma_bf16(a, bb, acc);
  }
  // --- epilogue: D-skip + SiLU(z) gate -> t ---
  const float Dh = Dp[h];
#pragma unroll
  for (int r = 0; r < 8; ++r) {
    int m = l0 + hf * 8 + r;
    size_t bl = blbase + m;
    int col = h * HEADDIM + p0 + lane16;
    float xv = xBC[bl * CONV_DIM + col];
    float zv = zx[bl * D_IN_PROJ + col];
    float y = acc[r] + Dh * xv;
    float sig = 1.f / (1.f + __expf(-zv));
    t[bl * D_SSM + col] = y * (zv * sig);
  }
}

// ---------------- 8: RMSNorm in place over D_SSM ----------------
__global__ void rmsnorm_kernel(float* __restrict__ t, const float* __restrict__ nw) {
  __shared__ float red[256];
  const size_t bl = blockIdx.x;
  float* row = t + bl * (size_t)D_SSM;
  float s = 0.f;
  for (int i = threadIdx.x; i < D_SSM; i += 256) { float v = row[i]; s += v * v; }
  red[threadIdx.x] = s;
  __syncthreads();
  for (int off = 128; off > 0; off >>= 1) {
    if (threadIdx.x < off) red[threadIdx.x] += red[threadIdx.x + off];
    __syncthreads();
  }
  const float r = rsqrtf(red[0] / (float)D_SSM + EPS_RMS);
  for (int i = threadIdx.x; i < D_SSM; i += 256) row[i] = row[i] * r * nw[i];
}

// ---------------- launcher ----------------
extern "C" void kernel_launch(void* const* d_in, const int* in_sizes, int n_in,
                              void* d_out, int out_size, void* d_ws, size_t ws_size,
                              hipStream_t stream) {
  const float* inputs  = (const float*)d_in[0];
  const float* W_in    = (const float*)d_in[1];
  const float* conv_w  = (const float*)d_in[2];
  const float* conv_b  = (const float*)d_in[3];
  const float* dt_bias = (const float*)d_in[4];
  const float* A_log   = (const float*)d_in[5];
  const float* Dvec    = (const float*)d_in[6];
  const float* norm_w  = (const float*)d_in[7];
  const float* W_out   = (const float*)d_in[8];
  float* out = (float*)d_out;
  float* ws  = (float*)d_ws;

  // workspace layout (floats); total ~91.4M floats (~366 MB)
  size_t o = 0;
  float* zx     = ws + o; o += (size_t)B_SZ * SEQ * D_IN_PROJ;
  float* xBC    = ws + o; o += (size_t)B_SZ * SEQ * CONV_DIM;
  float* dtp    = ws + o; o += (size_t)B_SZ * SEQ * NHEADS;
  float* cs     = ws + o; o += (size_t)B_SZ * NHEADS * SEQ;
  float* G      = ws + o; o += (size_t)B_SZ * NC * CHUNK * CHUNK;
  float* states = ws + o; o += (size_t)B_SZ * NC * NHEADS * HEADDIM * D_STATE;
  float* prev   = ws + o; o += (size_t)B_SZ * NC * NHEADS * HEADDIM * D_STATE;
  float* tbuf   = ws + o; o += (size_t)B_SZ * SEQ * D_SSM;

  const dim3 blk128(128, 1, 1);
  const dim3 w32(32, 1, 1);

  // 1. in_proj: zx[8192,4384] = inputs[8192,1024] @ W_in[4384,1024]^T
  gemm_bf16_lds<<<dim3(D_IN_PROJ / 16, (B_SZ * SEQ) / 256, 1), blk128, 0, stream>>>(
      inputs, D_MODEL, 0, W_in, D_MODEL, 0, zx, D_IN_PROJ, 0, D_MODEL);

  // 2. depthwise conv + SiLU + softplus(dt)
  conv_silu_dt<<<B_SZ * SEQ, 256, 0, stream>>>(zx, conv_w, conv_b, dt_bias, xBC, dtp);

  // 3. per-chunk cumsum of dA
  cumsum_dA<<<B_SZ * NHEADS * NC, CHUNK, 0, stream>>>(dtp, A_log, cs);

  // 4. G[b,z] = C[256,128] @ B[256,128]^T, batched over b*chunks
  gemm_bf16_lds<<<dim3(CHUNK / 16, CHUNK / 256, B_SZ * NC), blk128, 0, stream>>>(
      xBC + D_SSM + D_STATE, CONV_DIM, (long long)CHUNK * CONV_DIM,
      xBC + D_SSM,           CONV_DIM, (long long)CHUNK * CONV_DIM,
      G, CHUNK, (long long)CHUNK * CHUNK, D_STATE);

  // 5. chunk states
  states_kernel<<<dim3(D_STATE / 16, HEADDIM / 16, B_SZ * NC * NHEADS), w32, 0, stream>>>(
      xBC, dtp, cs, states);

  // 6. inter-chunk scan
  scan_prev<<<(B_SZ * NHEADS * HEADDIM * D_STATE) / 256, 256, 0, stream>>>(states, cs, prev);

  // 7. Y_diag + Y_off + D-skip + SiLU gate -> tbuf
  ydiag_kernel<<<dim3(HEADDIM / 16, CHUNK / 16, B_SZ * NC * NHEADS), w32, 0, stream>>>(
      xBC, zx, dtp, cs, G, prev, Dvec, tbuf);

  // 8. RMSNorm (in place)
  rmsnorm_kernel<<<B_SZ * SEQ, 256, 0, stream>>>(tbuf, norm_w);

  // 9. out_proj: out[8192,1024] = tbuf[8192,2048] @ W_out[1024,2048]^T
  gemm_bf16_lds<<<dim3(D_OUT / 16, (B_SZ * SEQ) / 256, 1), blk128, 0, stream>>>(
      tbuf, D_SSM, 0, W_out, D_SSM, 0, out, D_OUT, 0, D_SSM);

  (void)in_sizes; (void)n_in; (void)out_size; (void)ws_size;
}
